// MultiHeadAttention_12068858102084
// MI455X (gfx1250) — matrix operand order repaired and checked by reading, not compiled, not run on hardware
//
#include <hip/hip_runtime.h>

// MHA forward for MI455X (gfx1250, wave32, WMMA f16 16x16x32 with f32 accum).
// B=8, L=1024, D=1024, H=16, DK=64.

constexpr int BB  = 8;
constexpr int LL  = 1024;
constexpr int DD  = 1024;
constexpr int HH  = 16;
constexpr int DKK = 64;

typedef __attribute__((ext_vector_type(16))) _Float16 v16h;
typedef __attribute__((ext_vector_type(8)))  _Float16 v8h;
typedef __attribute__((ext_vector_type(8)))  float    v8f;

// ---- WMMA lane layouts (CDNA5 ISA 7.12.2) --------------------------------
// A 16x32 (MxK) 16-bit: lane m=lane&15, half=lane>>4.
//   a[0..7]  = row[half*16 + 0..7]        (contiguous 16B)
//   a[8..15] = row[half*16 + 8..15]   (contiguous 16B)
// B 32x16 (KxN) 16-bit: lane n=lane&15, half=lane>>4; VGPR v holds
//   K = half*16 + 2v (+0/1), N = n. A and B share this contiguous k order, not the ISA table order.
// C/D 16x16 f32: VGPR i, lane l -> M = i + 8*(l>>4), N = l&15.

// f32 source A-tile: convert in-register, fetch K-pairs as 8B loads.
__device__ inline v16h load_A_f32(const float* __restrict__ src, int ld, int lane) {
  const int m = lane & 15, half = lane >> 4;
  const float* p = src + (size_t)m * ld;
  v16h a;
#pragma unroll
  for (int v = 0; v < 8; ++v) {
    const int k = half * 16 + 2 * v;  // even -> 8B aligned
    const float2 t = *(const float2*)(p + k);
    a[2 * v]     = (_Float16)t.x;
    a[2 * v + 1] = (_Float16)t.y;
  }
  return a;
}

// f16 source A-tile: two 16B vector loads, no conversion.
__device__ inline v16h load_A_f16(const _Float16* __restrict__ src, int ld, int lane) {
  const int m = lane & 15, half = lane >> 4;
  const _Float16* p = src + (size_t)m * ld + half * 16;
  const v8h lo = *(const v8h*)(p);
  const v8h hi = *(const v8h*)(p + 8);
  return __builtin_shufflevector(lo, hi, 0, 1, 2, 3, 4, 5, 6, 7,
                                 8, 9, 10, 11, 12, 13, 14, 15);
}

// B from K-major f32 source: B[k][n] = src[k*ld + n] (e.g. Wo, Wq/Wk/Wv).
__device__ inline v16h load_B_km_f32(const float* __restrict__ src, int ld, int lane) {
  const int n = lane & 15, half = lane >> 4;
  v16h b;
#pragma unroll
  for (int v = 0; v < 8; ++v) {
    const int k = half * 16 + 2 * v;
    b[2 * v]     = (_Float16)src[(size_t)k * ld + n];
    b[2 * v + 1] = (_Float16)src[(size_t)(k + 1) * ld + n];
  }
  return b;
}

// B from K-major f16 source (vs): lanes coalesce across N, keep scalar per lane.
__device__ inline v16h load_B_km_f16(const _Float16* __restrict__ src, int ld, int lane) {
  const int n = lane & 15, half = lane >> 4;
  v16h b;
#pragma unroll
  for (int v = 0; v < 8; ++v) {
    const int k = half * 16 + 2 * v;
    b[2 * v]     = src[(size_t)k * ld + n];
    b[2 * v + 1] = src[(size_t)(k + 1) * ld + n];
  }
  return b;
}

// B from transposed f16 source: B[k][n] = src[n*ld + k]  (Kᵀ for scores).
// Per lane this is one contiguous 32B run -> two 16B vector loads.
__device__ inline v16h load_B_nm_f16(const _Float16* __restrict__ src, int ld, int lane) {
  const int n = lane & 15, half = lane >> 4;
  const _Float16* p = src + (size_t)n * ld + half * 16;
  const v8h lo = *(const v8h*)(p);
  const v8h hi = *(const v8h*)(p + 8);
  return __builtin_shufflevector(lo, hi, 0, 1, 2, 3, 4, 5, 6, 7,
                                 8, 9, 10, 11, 12, 13, 14, 15);
}

__device__ inline void store_C_f16(_Float16* __restrict__ dst, int ld, int lane,
                                   const v8f& c) {
  const int n = lane & 15, half = lane >> 4;
#pragma unroll
  for (int i = 0; i < 8; ++i) dst[(size_t)(i + 8 * half) * ld + n] = (_Float16)c[i];
}

#define WMMA_F16(a, b, c) \
  __builtin_amdgcn_wmma_f32_16x16x32_f16(false, (a), false, (b), (short)0, (c), false, false)

// ---- Kernel 1: per-head projection  X[B,L,D] x W[h][D,DK] -> f16 [B,H,L,DK]
__global__ __launch_bounds__(256) void proj_kernel(const float* __restrict__ X,
                                                   const float* __restrict__ W,
                                                   _Float16* __restrict__ O) {
  const int lane = threadIdx.x & 31;
  const int wave = threadIdx.x >> 5;
  const int tiles = LL / 128;
  const int t  = blockIdx.x % tiles;
  const int bh = blockIdx.x / tiles;
  const int b = bh / HH, h = bh % HH;
  const int l0 = t * 128 + wave * 16;

  const float* x = X + ((size_t)b * LL + l0) * DD;
  const float* w = W + (size_t)h * DD * DKK;

  v8f acc[4] = {};
  for (int kk = 0; kk < DD; kk += 32) {
    __builtin_prefetch(x + kk + 64, 0, 0);
    v16h a = load_A_f32(x + kk, DD, lane);
#pragma unroll
    for (int nt = 0; nt < 4; ++nt) {
      v16h bm = load_B_km_f32(w + (size_t)kk * DKK + nt * 16, DKK, lane);
      acc[nt] = WMMA_F16(a, bm, acc[nt]);
    }
  }
  _Float16* o = O + ((size_t)bh * LL + l0) * DKK;
#pragma unroll
  for (int nt = 0; nt < 4; ++nt) store_C_f16(o + nt * 16, DKK, lane, acc[nt]);
}

// ---- Kernel 2: scores = q̂ k̂ᵀ / 8 + mask, softmax, write f32 attn_cat -----
__global__ __launch_bounds__(256) void scores_softmax_kernel(
    const _Float16* __restrict__ qs, const _Float16* __restrict__ ks,
    const int* __restrict__ mask, float* __restrict__ attn) {
  __shared__ float s[16 * 1024];  // 64 KB score tile (LDS is 320 KB/WGP)

  const int lane = threadIdx.x & 31;
  const int wave = threadIdx.x >> 5;
  const int qt = blockIdx.x & (LL / 16 - 1);
  const int bh = blockIdx.x / (LL / 16);
  const int b = bh / HH, h = bh % HH;

  const _Float16* qrow = qs + ((size_t)bh * LL + qt * 16) * DKK;
  const _Float16* krow = ks + (size_t)bh * LL * DKK;

  v8f acc[8] = {};
#pragma unroll
  for (int kk = 0; kk < DKK; kk += 32) {
    v16h a = load_A_f16(qrow + kk, DKK, lane);
#pragma unroll
    for (int nt = 0; nt < 8; ++nt) {
      const int kcol = wave * 128 + nt * 16;
      v16h bm = load_B_nm_f16(krow + (size_t)kcol * DKK + kk, DKK, lane);
      acc[nt] = WMMA_F16(a, bm, acc[nt]);
    }
  }

  const int n = lane & 15, half = lane >> 4;
#pragma unroll
  for (int nt = 0; nt < 8; ++nt) {
#pragma unroll
    for (int i = 0; i < 8; ++i) {
      const int row = i + 8 * half;
      const int col = wave * 128 + nt * 16 + n;
      const int qq = qt * 16 + row;
      const float mv = mask[((size_t)b * LL + qq) * LL + col] ? 0.0f : -1e9f;
      s[row * 1024 + col] = acc[nt][i] * 0.125f + mv;
    }
  }
  __syncthreads();

  // Row softmax: 16 consecutive lanes per row (rows live inside one wave32).
  const int r  = threadIdx.x >> 4;
  const int c0 = threadIdx.x & 15;
  float mx = -3.0e38f;
  for (int c = c0; c < LL; c += 16) mx = fmaxf(mx, s[r * 1024 + c]);
#pragma unroll
  for (int off = 8; off > 0; off >>= 1) mx = fmaxf(mx, __shfl_xor(mx, off, 16));
  float sum = 0.0f;
  for (int c = c0; c < LL; c += 16) {
    const float e = __expf(s[r * 1024 + c] - mx);
    s[r * 1024 + c] = e;
    sum += e;
  }
#pragma unroll
  for (int off = 8; off > 0; off >>= 1) sum += __shfl_xor(sum, off, 16);
  const float inv = 1.0f / sum;

  const int qq = qt * 16 + r;
  float* arow = attn + ((size_t)b * LL + qq) * (size_t)(HH * LL) + (size_t)h * LL;
  for (int c = c0; c < LL; c += 16) arow[c] = s[r * 1024 + c] * inv;
}

// ---- Kernel 3: head = attn @ v̂  -> f16 head_cat [B,L,H*DK] ---------------
__global__ __launch_bounds__(256) void av_kernel(const float* __restrict__ attn,
                                                 const _Float16* __restrict__ vs,
                                                 _Float16* __restrict__ headcat) {
  const int lane = threadIdx.x & 31;
  const int wave = threadIdx.x >> 5;
  const int tiles = LL / 128;
  const int t  = blockIdx.x % tiles;
  const int bh = blockIdx.x / tiles;
  const int b = bh / HH, h = bh % HH;
  const int l0 = t * 128 + wave * 16;

  const float* arow = attn + ((size_t)b * LL + l0) * (size_t)(HH * LL) + (size_t)h * LL;
  const _Float16* v = vs + (size_t)bh * LL * DKK;

  v8f acc[4] = {};
  for (int kk = 0; kk < LL; kk += 32) {
    v16h a = load_A_f32(arow + kk, HH * LL, lane);  // f32 probs -> f16 in-register
#pragma unroll
    for (int nt = 0; nt < 4; ++nt) {
      v16h bm = load_B_km_f16(v + (size_t)kk * DKK + nt * 16, DKK, lane);
      acc[nt] = WMMA_F16(a, bm, acc[nt]);
    }
  }
  _Float16* o = headcat + ((size_t)b * LL + l0) * DD + h * DKK;
#pragma unroll
  for (int nt = 0; nt < 4; ++nt) store_C_f16(o + nt * 16, DD, lane, acc[nt]);
}

// ---- Kernel 4: out = head_cat @ Wo + bo  (f32 out) ------------------------
__global__ __launch_bounds__(256) void outproj_kernel(const _Float16* __restrict__ hc,
                                                      const float* __restrict__ Wo,
                                                      const float* __restrict__ bo,
                                                      float* __restrict__ out) {
  const int lane = threadIdx.x & 31;
  const int wave = threadIdx.x >> 5;
  const int m0 = blockIdx.x * 128 + wave * 16;  // rows of B*L
  const int n0 = blockIdx.y * 64;

  const _Float16* x = hc + (size_t)m0 * DD;
  v8f acc[4] = {};
  for (int kk = 0; kk < DD; kk += 32) {
    v16h a = load_A_f16(x + kk, DD, lane);
#pragma unroll
    for (int nt = 0; nt < 4; ++nt) {
      v16h bm = load_B_km_f32(Wo + (size_t)kk * DD + n0 + nt * 16, DD, lane);
      acc[nt] = WMMA_F16(a, bm, acc[nt]);
    }
  }
  const int n = lane & 15, half = lane >> 4;
#pragma unroll
  for (int nt = 0; nt < 4; ++nt) {
    const int col = n0 + nt * 16 + n;
    const float bias = bo[col];
#pragma unroll
    for (int i = 0; i < 8; ++i)
      out[(size_t)(m0 + i + 8 * half) * DD + col] = acc[nt][i] + bias;
  }
}

extern "C" void kernel_launch(void* const* d_in, const int* in_sizes, int n_in,
                              void* d_out, int out_size, void* d_ws, size_t ws_size,
                              hipStream_t stream) {
  (void)in_sizes; (void)n_in; (void)out_size; (void)ws_size;
  const float* q    = (const float*)d_in[0];
  const float* k    = (const float*)d_in[1];
  const float* v    = (const float*)d_in[2];
  const int*   mask = (const int*)d_in[3];
  const float* Wq   = (const float*)d_in[4];
  const float* Wk   = (const float*)d_in[5];
  const float* Wv   = (const float*)d_in[6];
  const float* Wo   = (const float*)d_in[7];
  const float* bo   = (const float*)d_in[8];

  float* out  = (float*)d_out;                                 // [B,L,D]
  float* attn = out + (size_t)BB * LL * DD;                    // [B,L,H*L]

  const size_t headElems = (size_t)BB * HH * LL * DKK;         // 8M f16 each
  _Float16* qs = (_Float16*)d_ws;
  _Float16* ks = qs + headElems;
  _Float16* vs = ks + headElems;
  _Float16* hc = vs + headElems;                               // [B,L,H*DK] f16

  const dim3 blk(256);
  const dim3 gProj(BB * HH * (LL / 128));                      // 1024 blocks
  proj_kernel<<<gProj, blk, 0, stream>>>(q, Wq, qs);
  proj_kernel<<<gProj, blk, 0, stream>>>(k, Wk, ks);
  proj_kernel<<<gProj, blk, 0, stream>>>(v, Wv, vs);

  const dim3 gAttn(BB * HH * (LL / 16));                       // 8192 blocks
  scores_softmax_kernel<<<gAttn, blk, 0, stream>>>(qs, ks, mask, attn);

  av_kernel<<<gProj, blk, 0, stream>>>(attn, vs, hc);

  const dim3 gOut((BB * LL) / 128, DD / 64);                   // 64 x 16
  outproj_kernel<<<gOut, blk, 0, stream>>>(hc, Wo, bo, out);
}
